// Transformer_68839735820666
// MI455X (gfx1250) — matrix operand order repaired and checked by reading, not compiled, not run on hardware
//
#include <hip/hip_runtime.h>
#include <hip/hip_bf16.h>
#include <stdint.h>

typedef __attribute__((ext_vector_type(16))) _Float16 v16h;
typedef __attribute__((ext_vector_type(8)))  _Float16 h8;
typedef __attribute__((ext_vector_type(8)))  float    v8f;

#define D_    256
#define NH_   8
#define HD_   32
#define DFF_  1024
#define BS_   4
#define NQ_   900
#define LV_   20197
#define M1_   (BS_*NQ_)      // 3600
#define MV_   (BS_*LV_)      // 80788
#define SLD_  928            // padded attention row length (multiple of 32)

// ---------------------------------------------------------------------------
// Generic batched WMMA GEMM:  C[m,n] = sum_k A[m,k] * B[n,k]  (+bias, +relu)
// A: (M,K) row-major f16, lda mult of 8.  B: (N,K) row-major f16, ldb mult 16.
// K must be a multiple of 32. One wave32 computes a 16x16 C tile using
// v_wmma_f32_16x16x32_f16; fragments loaded per CDNA5 ISA 7.12.2 layouts.
// ---------------------------------------------------------------------------
__global__ __launch_bounds__(128)
void wmma_gemm_kernel(const _Float16* __restrict__ A, const _Float16* __restrict__ B,
                      const float* __restrict__ bias, float* __restrict__ Cf,
                      _Float16* __restrict__ Ch,
                      int M, int N, int K, int lda, int ldb, int ldc,
                      long long sAo, long long sAi, long long sBo, long long sBi,
                      long long sCo, long long sCi, int nInner, int relu)
{
    int z = blockIdx.z;
    int outer = z / nInner, inner = z - outer * nInner;
    A += outer * sAo + inner * sAi;
    B += outer * sBo + inner * sBi;
    long long cBase = outer * sCo + inner * sCi;

    int lane = threadIdx.x & 31;
    int wave = threadIdx.x >> 5;
    int m0 = (blockIdx.y * 4 + wave) * 16;
    int n0 = blockIdx.x * 16;
    if (m0 >= M) return;                       // wave-uniform exit

    int r = m0 + (lane & 15); if (r > M - 1) r = M - 1;   // clamp edge rows
    int n = n0 + (lane & 15); if (n > N - 1) n = N - 1;   // clamp edge cols
    int hh = lane >> 4;

    // A frag (16x32 f16): lane row = lane%16; halves hold K runs h*8..h*8+7
    // and 16+h*8..16+h*8+7 -> two contiguous 16B loads per k-step.
    const _Float16* ap = A + (long long)r * lda + hh * 8;
    // B frag (32x16 f16): lane col = lane%16; halves hold K runs h*8..h*8+7
    // and 16+h*8..16+h*8+7 -> two contiguous 16B loads per k-step.
    const _Float16* bp = B + (long long)n * ldb + hh * 8;

    v8f acc = {};
    for (int k0 = 0; k0 < K; k0 += 32) {
        union { v16h v; h8 h[2]; } a;
        a.h[0] = *(const h8*)(ap + k0);
        a.h[1] = *(const h8*)(ap + k0 + 16);
        union { v16h v; h8 h[2]; } b;
        b.h[0] = *(const h8*)(bp + k0);
        b.h[1] = *(const h8*)(bp + k0 + 16);
        __builtin_prefetch(ap + k0 + 64, 0, 1);
        acc = __builtin_amdgcn_wmma_f32_16x16x32_f16(
            false, a.v, false, b.v, (short)0, acc, false, false);
    }

    int nc = n0 + (lane & 15);
    if (nc < N) {
        float bv = bias ? bias[nc] : 0.0f;
        #pragma unroll
        for (int i = 0; i < 8; ++i) {          // c[i] = C[h*8+i][lane%16]
            int m = m0 + hh * 8 + i;
            if (m < M) {
                float v = acc[i] + bv;
                if (relu) v = fmaxf(v, 0.0f);
                long long idx = cBase + (long long)m * ldc + nc;
                if (Cf) Cf[idx] = v;
                if (Ch) Ch[idx] = (_Float16)v;
            }
        }
    }
}

// ---------------------------------------------------------------------------
__global__ void cvt_kernel(const float* __restrict__ in, _Float16* __restrict__ out,
                           long long n) {
    long long i = (long long)blockIdx.x * blockDim.x + threadIdx.x;
    if (i < n) out[i] = (_Float16)in[i];
}

__global__ void add_cvt_kernel(const float* __restrict__ a, const float* __restrict__ b,
                               _Float16* __restrict__ out, long long n) {
    long long i = (long long)blockIdx.x * blockDim.x + threadIdx.x;
    if (i < n) out[i] = (_Float16)(a[i] + b[i]);
}

// zero padded-value rows where key_padding_mask is set
__global__ void mask_zero_kernel(const unsigned char* __restrict__ mask,
                                 _Float16* __restrict__ v) {
    int row = blockIdx.x;
    if (!mask[row]) return;
    _Float16* p = v + (long long)row * D_;
    for (int i = threadIdx.x; i < D_; i += blockDim.x) p[i] = (_Float16)0.0f;
}

// softmax over the 16 (level,point) attention logits per (b,q,head)
__global__ void aw_softmax_kernel(const float* __restrict__ logit,
                                  float* __restrict__ prob) {
    int t = blockIdx.x * blockDim.x + threadIdx.x;
    if (t >= M1_ * NH_) return;
    int m = t >> 3, h = t & 7;
    const float* in = logit + (long long)m * 128 + h * 16;
    float* out = prob + (long long)m * 128 + h * 16;
    float mx = -3.0e38f;
    for (int i = 0; i < 16; ++i) mx = fmaxf(mx, in[i]);
    float e[16], s = 0.f;
    for (int i = 0; i < 16; ++i) { e[i] = __expf(in[i] - mx); s += e[i]; }
    float inv = 1.0f / s;
    for (int i = 0; i < 16; ++i) out[i] = e[i] * inv;
}

// MS-deform bilinear sampling: 1 wave per (b,q,head), lane = channel.
// Gathers are 32-lane contiguous (64B) per corner -> coalesced.
__global__ void msdeform_sample_kernel(const float* __restrict__ off,
                                       const float* __restrict__ aw,
                                       const float* __restrict__ ref,
                                       const _Float16* __restrict__ v,
                                       _Float16* __restrict__ out)
{
    int m = blockIdx.x;                      // b*NQ + q
    int b = m / NQ_;
    int h = threadIdx.x >> 5;
    int c = threadIdx.x & 31;
    const int Hs[4] = {100, 50, 25, 13};
    const int Ws[4] = {152, 76, 38, 19};
    const int st[4] = {0, 15200, 19000, 19950};
    const float* offr = off + (long long)m * D_ + h * 32;
    const float* awr  = aw  + (long long)m * 128 + h * 16;
    const float* refr = ref + (long long)m * 8;
    const _Float16* vb = v + (long long)b * LV_ * D_ + h * HD_ + c;
    float acc = 0.f;
    #pragma unroll
    for (int l = 0; l < 4; ++l) {
        int Hh = Hs[l], Ww = Ws[l];
        float fW = (float)Ww, fH = (float)Hh;
        float rx = refr[l * 2 + 0], ry = refr[l * 2 + 1];
        const _Float16* vl = vb + (long long)st[l] * D_;
        #pragma unroll
        for (int p = 0; p < 4; ++p) {
            float ox = offr[(l * 4 + p) * 2 + 0];
            float oy = offr[(l * 4 + p) * 2 + 1];
            float x = (rx + ox / fW) * fW - 0.5f;
            float y = (ry + oy / fH) * fH - 0.5f;
            float x0f = floorf(x), y0f = floorf(y);
            float fx = x - x0f, fy = y - y0f;
            int x0 = (int)x0f, y0 = (int)y0f;
            bool xi0 = (x0 >= 0) && (x0 < Ww), xi1 = (x0 + 1 >= 0) && (x0 + 1 < Ww);
            bool yi0 = (y0 >= 0) && (y0 < Hh), yi1 = (y0 + 1 >= 0) && (y0 + 1 < Hh);
            float g00 = 0.f, g01 = 0.f, g10 = 0.f, g11 = 0.f;
            if (yi0 && xi0) g00 = (float)vl[(long long)(y0 * Ww + x0) * D_];
            if (yi0 && xi1) g01 = (float)vl[(long long)(y0 * Ww + x0 + 1) * D_];
            if (yi1 && xi0) g10 = (float)vl[(long long)((y0 + 1) * Ww + x0) * D_];
            if (yi1 && xi1) g11 = (float)vl[(long long)((y0 + 1) * Ww + x0 + 1) * D_];
            float s = (g00 * (1.f - fx) + g01 * fx) * (1.f - fy)
                    + (g10 * (1.f - fx) + g11 * fx) * fy;
            acc += awr[l * 4 + p] * s;
        }
    }
    out[(long long)m * D_ + h * HD_ + c] = (_Float16)acc;
}

// residual-add + LayerNorm; optionally emits f16 copy and f16(x+pos)
__global__ void add_ln_kernel(const float* __restrict__ x, const float* __restrict__ y,
                              const float* __restrict__ g, const float* __restrict__ bt,
                              float* __restrict__ out, _Float16* __restrict__ out16,
                              const float* __restrict__ pos,
                              _Float16* __restrict__ outpos16)
{
    __shared__ float red[D_];
    int row = blockIdx.x, c = threadIdx.x;
    long long idx = (long long)row * D_ + c;
    float s = x[idx] + y[idx];
    red[c] = s; __syncthreads();
    for (int o = D_ / 2; o > 0; o >>= 1) {
        if (c < o) red[c] += red[c + o];
        __syncthreads();
    }
    float mean = red[0] * (1.0f / D_);
    __syncthreads();
    float d = s - mean;
    red[c] = d * d; __syncthreads();
    for (int o = D_ / 2; o > 0; o >>= 1) {
        if (c < o) red[c] += red[c + o];
        __syncthreads();
    }
    float var = red[0] * (1.0f / D_);
    float ny = d * rsqrtf(var + 1e-5f) * g[c] + bt[c];
    out[idx] = ny;
    if (out16) out16[idx] = (_Float16)ny;
    if (outpos16) outpos16[idx] = (_Float16)(ny + pos[idx]);
}

// row softmax (900 valid cols, padded to 928 with zeros), scale folded in
__global__ void mha_softmax_kernel(_Float16* __restrict__ sc) {
    long long row = blockIdx.x;
    _Float16* p = sc + row * (long long)SLD_;
    __shared__ float red[128];
    int t = threadIdx.x;
    const float scale = 0.17677669529663687f;   // 1/sqrt(HD)
    float mx = -3.0e38f;
    for (int i = t; i < NQ_; i += 128) mx = fmaxf(mx, (float)p[i] * scale);
    red[t] = mx; __syncthreads();
    for (int o = 64; o > 0; o >>= 1) { if (t < o) red[t] = fmaxf(red[t], red[t + o]); __syncthreads(); }
    mx = red[0]; __syncthreads();
    float s = 0.f;
    for (int i = t; i < NQ_; i += 128) {
        float e = __expf((float)p[i] * scale - mx);
        s += e;
        p[i] = (_Float16)e;
    }
    red[t] = s; __syncthreads();
    for (int o = 64; o > 0; o >>= 1) { if (t < o) red[t] += red[t + o]; __syncthreads(); }
    float inv = 1.0f / red[0];
    __syncthreads();
    for (int i = t; i < SLD_; i += 128)
        p[i] = (i < NQ_) ? (_Float16)((float)p[i] * inv) : (_Float16)0.0f;
}

// V-head transpose: vm(b*NQ+key, h*32+c) f16 -> vT[(b,h,c), key] padded to 928
__global__ void vt_kernel(const _Float16* __restrict__ vm, _Float16* __restrict__ vt) {
    long long o = (long long)blockIdx.x * blockDim.x + threadIdx.x;
    const long long total = (long long)BS_ * NH_ * HD_ * SLD_;
    if (o >= total) return;
    int key = (int)(o % SLD_);
    long long rest = o / SLD_;
    int c = (int)(rest % HD_);
    int bh = (int)(rest / HD_);
    int b = bh >> 3, h = bh & 7;
    _Float16 v = (_Float16)0.0f;
    if (key < NQ_) v = vm[(long long)(b * NQ_ + key) * D_ + h * HD_ + c];
    vt[o] = v;
}

// ---------------------------------------------------------------------------
extern "C" void kernel_launch(void* const* d_in, const int* in_sizes, int n_in,
                              void* d_out, int out_size, void* d_ws, size_t ws_size,
                              hipStream_t stream)
{
    (void)in_sizes; (void)n_in; (void)out_size; (void)ws_size;
    // inputs in setup_inputs() insertion order (params nested dict flattened in order)
    const float* tgt   = (const float*)d_in[0];
    const float* pos   = (const float*)d_in[1];
    const float* ref   = (const float*)d_in[2];
    const float* mem   = (const float*)d_in[3];
    const unsigned char* mask = (const unsigned char*)d_in[4];
    const float* value_w = (const float*)d_in[5];
    const float* value_b = (const float*)d_in[6];
    const float* soff_w  = (const float*)d_in[7];
    const float* soff_b  = (const float*)d_in[8];
    const float* attw_w  = (const float*)d_in[9];
    const float* attw_b  = (const float*)d_in[10];
    const float* outp_w  = (const float*)d_in[11];
    const float* outp_b  = (const float*)d_in[12];
    const float* inpj_w  = (const float*)d_in[13];
    const float* inpj_b  = (const float*)d_in[14];
    const float* mhao_w  = (const float*)d_in[15];
    const float* mhao_b  = (const float*)d_in[16];
    const float* ln1g = (const float*)d_in[17];
    const float* ln1b = (const float*)d_in[18];
    const float* ln2g = (const float*)d_in[19];
    const float* ln2b = (const float*)d_in[20];
    const float* ln3g = (const float*)d_in[21];
    const float* ln3b = (const float*)d_in[22];
    const float* lin1_w = (const float*)d_in[23];
    const float* lin1_b = (const float*)d_in[24];
    const float* lin2_w = (const float*)d_in[25];
    const float* lin2_b = (const float*)d_in[26];

    // bump allocator over d_ws (~190 MB total)
    uintptr_t base = (uintptr_t)d_ws;
    size_t off = 0;
    auto alloc = [&](size_t bytes) -> void* {
        void* p = (void*)(base + off);
        off = (off + bytes + 255) & ~(size_t)255;
        return p;
    };
    _Float16* value_w16 = (_Float16*)alloc((size_t)65536 * 2);
    _Float16* soff_w16  = (_Float16*)alloc((size_t)65536 * 2);
    _Float16* attw_w16  = (_Float16*)alloc((size_t)32768 * 2);
    _Float16* outp_w16  = (_Float16*)alloc((size_t)65536 * 2);
    _Float16* inpj_w16  = (_Float16*)alloc((size_t)196608 * 2);
    _Float16* mhao_w16  = (_Float16*)alloc((size_t)65536 * 2);
    _Float16* lin1_w16  = (_Float16*)alloc((size_t)262144 * 2);
    _Float16* lin2_w16  = (_Float16*)alloc((size_t)262144 * 2);
    _Float16* q16       = (_Float16*)alloc((size_t)M1_ * D_ * 2);
    _Float16* mem16     = (_Float16*)alloc((size_t)MV_ * D_ * 2);
    _Float16* vproj16   = (_Float16*)alloc((size_t)MV_ * D_ * 2);
    float*    offbuf    = (float*)alloc((size_t)M1_ * D_ * 4);
    float*    awlog     = (float*)alloc((size_t)M1_ * 128 * 4);
    float*    awp       = (float*)alloc((size_t)M1_ * 128 * 4);
    _Float16* msout16   = (_Float16*)alloc((size_t)M1_ * D_ * 2);
    float*    tgt2ms    = (float*)alloc((size_t)M1_ * D_ * 4);
    float*    tgtA      = (float*)alloc((size_t)M1_ * D_ * 4);
    _Float16* tgtA16    = (_Float16*)alloc((size_t)M1_ * D_ * 2);
    _Float16* qkpos16   = (_Float16*)alloc((size_t)M1_ * D_ * 2);
    _Float16* qkproj16  = (_Float16*)alloc((size_t)M1_ * 512 * 2);
    _Float16* vm16      = (_Float16*)alloc((size_t)M1_ * D_ * 2);
    _Float16* vT16      = (_Float16*)alloc((size_t)BS_ * NH_ * HD_ * SLD_ * 2);
    _Float16* scores16  = (_Float16*)alloc((size_t)BS_ * NH_ * NQ_ * SLD_ * 2);
    _Float16* attno16   = (_Float16*)alloc((size_t)M1_ * D_ * 2);
    float*    mha2      = (float*)alloc((size_t)M1_ * D_ * 4);
    float*    tgtB      = (float*)alloc((size_t)M1_ * D_ * 4);
    _Float16* tgtB16    = (_Float16*)alloc((size_t)M1_ * D_ * 2);
    _Float16* hff16     = (_Float16*)alloc((size_t)M1_ * DFF_ * 2);
    float*    ffn       = (float*)alloc((size_t)M1_ * D_ * 4);

    auto cvt = [&](const float* in, _Float16* o, long long n) {
        cvt_kernel<<<dim3((unsigned)((n + 255) / 256)), 256, 0, stream>>>(in, o, n);
    };
    auto gemm = [&](const _Float16* A, const _Float16* B, const float* bias,
                    float* Cf, _Float16* Ch, int M, int N, int K,
                    int lda, int ldb, int ldc, int batch, int nInner,
                    long long sAo, long long sAi, long long sBo, long long sBi,
                    long long sCo, long long sCi, int relu) {
        dim3 g((N + 15) / 16, (M + 63) / 64, batch);
        wmma_gemm_kernel<<<g, 128, 0, stream>>>(A, B, bias, Cf, Ch, M, N, K,
                                                lda, ldb, ldc, sAo, sAi, sBo, sBi,
                                                sCo, sCi, nInner, relu);
    };

    // --- f16 staging ---
    cvt(value_w, value_w16, 65536);  cvt(soff_w, soff_w16, 65536);
    cvt(attw_w, attw_w16, 32768);    cvt(outp_w, outp_w16, 65536);
    cvt(inpj_w, inpj_w16, 196608);   cvt(mhao_w, mhao_w16, 65536);
    cvt(lin1_w, lin1_w16, 262144);   cvt(lin2_w, lin2_w16, 262144);
    cvt(mem, mem16, (long long)MV_ * D_);
    add_cvt_kernel<<<dim3((M1_ * D_ + 255) / 256), 256, 0, stream>>>(
        tgt, pos, q16, (long long)M1_ * D_);

    // --- MS-deform attention ---
    gemm(mem16, value_w16, value_b, nullptr, vproj16, MV_, D_, D_, D_, D_, D_,
         1, 1, 0, 0, 0, 0, 0, 0, 0);
    mask_zero_kernel<<<dim3(MV_), 64, 0, stream>>>(mask, vproj16);
    gemm(q16, soff_w16, soff_b, offbuf, nullptr, M1_, D_, D_, D_, D_, D_,
         1, 1, 0, 0, 0, 0, 0, 0, 0);
    gemm(q16, attw_w16, attw_b, awlog, nullptr, M1_, 128, D_, D_, D_, 128,
         1, 1, 0, 0, 0, 0, 0, 0, 0);
    aw_softmax_kernel<<<dim3((M1_ * NH_ + 255) / 256), 256, 0, stream>>>(awlog, awp);
    msdeform_sample_kernel<<<dim3(M1_), 256, 0, stream>>>(offbuf, awp, ref, vproj16, msout16);
    gemm(msout16, outp_w16, outp_b, tgt2ms, nullptr, M1_, D_, D_, D_, D_, D_,
         1, 1, 0, 0, 0, 0, 0, 0, 0);
    add_ln_kernel<<<dim3(M1_), D_, 0, stream>>>(tgt, tgt2ms, ln1g, ln1b,
                                                tgtA, tgtA16, pos, qkpos16);

    // --- self-attention ---
    gemm(qkpos16, inpj_w16, inpj_b, nullptr, qkproj16, M1_, 512, D_, D_, D_, 512,
         1, 1, 0, 0, 0, 0, 0, 0, 0);                         // Q|K projection
    gemm(tgtA16, inpj_w16 + (size_t)512 * D_, inpj_b + 512, nullptr, vm16,
         M1_, D_, D_, D_, D_, D_, 1, 1, 0, 0, 0, 0, 0, 0, 0); // V projection
    {
        long long total = (long long)BS_ * NH_ * HD_ * SLD_;
        vt_kernel<<<dim3((unsigned)((total + 255) / 256)), 256, 0, stream>>>(vm16, vT16);
    }
    // scores: per (b,h) 900x900, K=32
    gemm(qkproj16, qkproj16 + 256, nullptr, nullptr, scores16,
         NQ_, NQ_, 32, 512, 512, SLD_, BS_ * NH_, NH_,
         (long long)NQ_ * 512, 32, (long long)NQ_ * 512, 32,
         (long long)NH_ * NQ_ * SLD_, (long long)NQ_ * SLD_, 0);
    mha_softmax_kernel<<<dim3(BS_ * NH_ * NQ_), 128, 0, stream>>>(scores16);
    // P @ V: per (b,h) 900x32, K=928 (zero-padded)
    gemm(scores16, vT16, nullptr, nullptr, attno16,
         NQ_, HD_, SLD_, SLD_, SLD_, D_, BS_ * NH_, NH_,
         (long long)NH_ * NQ_ * SLD_, (long long)NQ_ * SLD_,
         (long long)NH_ * HD_ * SLD_, (long long)HD_ * SLD_,
         (long long)NQ_ * D_, HD_, 0);
    gemm(attno16, mhao_w16, mhao_b, mha2, nullptr, M1_, D_, D_, D_, D_, D_,
         1, 1, 0, 0, 0, 0, 0, 0, 0);
    add_ln_kernel<<<dim3(M1_), D_, 0, stream>>>(tgtA, mha2, ln2g, ln2b,
                                                tgtB, tgtB16, nullptr, nullptr);

    // --- FFN ---
    gemm(tgtB16, lin1_w16, lin1_b, nullptr, hff16, M1_, DFF_, D_, D_, D_, DFF_,
         1, 1, 0, 0, 0, 0, 0, 0, 1);                         // +ReLU
    gemm(hff16, lin2_w16, lin2_b, ffn, nullptr, M1_, D_, DFF_, DFF_, DFF_, D_,
         1, 1, 0, 0, 0, 0, 0, 0, 0);
    add_ln_kernel<<<dim3(M1_), D_, 0, stream>>>(tgtB, ffn, ln3g, ln3b,
                                                (float*)d_out, nullptr, nullptr, nullptr);
}
